// Graph_Transformer_48146583388261
// MI455X (gfx1250) — compile-verified, matrix operand order checked by reading, not run on hardware
//
#include <hip/hip_runtime.h>
#include <hip/hip_bf16.h>

// ---------------------------------------------------------------------------
// Graph TransformerConv x2 + Linear head for MI455X (gfx1250, wave32, WMMA).
// GEMMs use v_wmma_f32_16x16x32_bf16 (bf16 A/B, f32 accumulate).
// Edge softmax/aggregation uses sortable-uint atomicMax + f32 atomicAdd.
// ---------------------------------------------------------------------------

typedef __attribute__((ext_vector_type(16))) __bf16 v16bf;
typedef __attribute__((ext_vector_type(8)))  float  v8f;
typedef __attribute__((ext_vector_type(4)))  unsigned int u32x4;

#define N_NODES 50000
#define N_EDGES 800000
#define HEADS   8

// ---- helpers --------------------------------------------------------------

__device__ inline __bf16 f2bf(float f) {
    unsigned u = __float_as_uint(f);
    u += 0x7FFFu + ((u >> 16) & 1u);           // round-to-nearest-even
    unsigned short hs = (unsigned short)(u >> 16);
    return __builtin_bit_cast(__bf16, hs);
}

// order-preserving float <-> uint mapping (for integer atomicMax on floats)
__device__ inline unsigned f2sortable(float f) {
    unsigned u = __float_as_uint(f);
    return (u & 0x80000000u) ? ~u : (u | 0x80000000u);
}
__device__ inline float sortable2f(unsigned u) {
    return (u & 0x80000000u) ? __uint_as_float(u & 0x7FFFFFFFu)
                             : __uint_as_float(~u);
}

// ---- conversion / packing kernels -----------------------------------------

__global__ void k_cvt_f32_to_bf16(const float* __restrict__ src,
                                  __bf16* __restrict__ dst, int n) {
    int t = blockIdx.x * blockDim.x + threadIdx.x;
    if (t < n) dst[t] = f2bf(src[t]);
}

// pack 4 weight matrices (K x Nc each, row-major) transposed into
// Wt[4*Nc][K] bf16, and pack biases into bpack[4*Nc]
__global__ void k_pack_weights(const float* __restrict__ Wq, const float* __restrict__ Wk,
                               const float* __restrict__ Wv, const float* __restrict__ Ws,
                               const float* __restrict__ bq, const float* __restrict__ bk,
                               const float* __restrict__ bv, const float* __restrict__ bs,
                               __bf16* __restrict__ Wt, float* __restrict__ bpack,
                               int K, int Nc) {
    int total = 4 * Nc * K;
    int t = blockIdx.x * blockDim.x + threadIdx.x;
    if (t >= total) return;
    int n = t / K;            // packed output column 0..4*Nc-1
    int k = t - n * K;
    int mat = n / Nc;
    int nn  = n - mat * Nc;
    const float* W = (mat == 0) ? Wq : (mat == 1) ? Wk : (mat == 2) ? Wv : Ws;
    Wt[(size_t)n * K + k] = f2bf(W[(size_t)k * Nc + nn]);
    if (k == 0) {
        const float* b = (mat == 0) ? bq : (mat == 1) ? bk : (mat == 2) ? bv : bs;
        bpack[n] = b[nn];
    }
}

// ---- WMMA GEMM:  C[M x 4*NCOLS] = A[M x K] @ Bt^T + bias, split to 4 bufs --
// A: bf16 row-major [M][K];  Bt: bf16 [4*NCOLS][K] (transposed weights)
// Each wave computes one 16x16 tile. K-loop in steps of 32 (bf16 WMMA).

union Frag { v16bf v; u32x4 u[2]; };

template<int NCOLS>   // columns per output buffer (128 for conv1, 8 for conv2)
__global__ void k_wmma_gemm(const __bf16* __restrict__ A,
                            const __bf16* __restrict__ Bt,
                            const float*  __restrict__ biasPacked,
                            float* __restrict__ out0, float* __restrict__ out1,
                            float* __restrict__ out2, float* __restrict__ out3,
                            int K) {
    const int mtile = blockIdx.x;
    const int ntile = blockIdx.y;
    const int lane  = threadIdx.x & 31;
    const int hf    = lane >> 4;      // 0 or 1
    const int lr    = lane & 15;
    const int m0    = mtile * 16;
    const int ncol  = ntile * 16 + lr;

    const __bf16* arow = A  + (size_t)(m0 + lr) * K;
    const __bf16* brow = Bt + (size_t)ncol      * K;

    v8f acc = {};
    for (int k0 = 0; k0 < K; k0 += 32) {
        Frag fa, fb;
        // A fragment: row (m0+lr); K runs [k0+hf*8, +8) and [k0+16+hf*8, +8)
        fa.u[0] = *(const u32x4*)(arow + k0 + hf * 8);
        fa.u[1] = *(const u32x4*)(arow + k0 + 16 + hf * 8);
        // B fragment: column ncol of W == row ncol of Bt, same K pattern
        fb.u[0] = *(const u32x4*)(brow + k0 + hf * 8);
        fb.u[1] = *(const u32x4*)(brow + k0 + 16 + hf * 8);
        acc = __builtin_amdgcn_wmma_f32_16x16x32_bf16(
                  false, fa.v, false, fb.v, (short)0, acc, false, false);
    }

    // epilogue: all 8 accum elements of a lane share the same output column
    const float bias = biasPacked[ncol];
    const int buf = ncol / NCOLS;
    const int cw  = ncol - buf * NCOLS;
    float* const outs[4] = { out0, out1, out2, out3 };
    float* op = outs[buf];
    #pragma unroll
    for (int r = 0; r < 8; ++r) {
        int m = m0 + r + 8 * hf;     // C layout: VGPR r -> row r (+8 for hi half)
        op[(size_t)m * NCOLS + cw] = acc[r] + bias;
    }
}

// ---- edge-softmax pipeline -------------------------------------------------

__global__ void k_init_softmax(unsigned* __restrict__ maxbuf,
                               float* __restrict__ sumbuf, int n) {
    int t = blockIdx.x * blockDim.x + threadIdx.x;
    if (t < n) { maxbuf[t] = 0u; sumbuf[t] = 0.0f; }
}

template<int C>
__global__ void k_score_max(const int* __restrict__ ei,
                            const float* __restrict__ Q, const float* __restrict__ K,
                            float* __restrict__ scores, unsigned* __restrict__ maxbuf,
                            float scale) {
    int t = blockIdx.x * blockDim.x + threadIdx.x;
    if (t >= N_EDGES * HEADS) return;
    int e = t >> 3, h = t & 7;
    int s = ei[e];
    int d = ei[N_EDGES + e];
    const float* q = Q + (size_t)d * HEADS * C + h * C;
    const float* k = K + (size_t)s * HEADS * C + h * C;
    float dot;
    if constexpr (C == 1) {
        dot = q[0] * k[0];
    } else {
        dot = 0.0f;
        #pragma unroll
        for (int c = 0; c < C; c += 4) {
            float4 qa = *(const float4*)(q + c);
            float4 ka = *(const float4*)(k + c);
            dot += qa.x * ka.x + qa.y * ka.y + qa.z * ka.z + qa.w * ka.w;
        }
    }
    float sc = dot * scale;
    scores[t] = sc;
    atomicMax(maxbuf + (size_t)d * HEADS + h, f2sortable(sc));
}

__global__ void k_exp_sum(const int* __restrict__ ei,
                          const unsigned* __restrict__ maxbuf,
                          float* __restrict__ scores, float* __restrict__ sumbuf) {
    int t = blockIdx.x * blockDim.x + threadIdx.x;
    if (t >= N_EDGES * HEADS) return;
    int e = t >> 3, h = t & 7;
    int d = ei[N_EDGES + e];
    float m  = sortable2f(maxbuf[(size_t)d * HEADS + h]);
    float ev = __expf(scores[t] - m);
    scores[t] = ev;
    atomicAdd(sumbuf + (size_t)d * HEADS + h, ev);
}

// Out must already contain the skip connection (x @ Ws + bs)
template<int C>
__global__ void k_aggregate(const int* __restrict__ ei,
                            const float* __restrict__ scores,
                            const float* __restrict__ sumbuf,
                            const float* __restrict__ V,
                            float* __restrict__ Out) {
    int t = blockIdx.x * blockDim.x + threadIdx.x;
    if (t >= N_EDGES * HEADS) return;
    int e = t >> 3, h = t & 7;
    int s = ei[e];
    int d = ei[N_EDGES + e];
    float alpha = scores[t] / (sumbuf[(size_t)d * HEADS + h] + 1e-16f);
    const float* v = V   + (size_t)s * HEADS * C + h * C;
    float*       o = Out + (size_t)d * HEADS * C + h * C;
    #pragma unroll
    for (int c = 0; c < C; ++c)
        atomicAdd(o + c, v[c] * alpha);
}

// ---- final head: mean_n relu(H2 @ Wl + bl) --------------------------------

__global__ void k_zero_out(float* __restrict__ out, int n) {
    int t = blockIdx.x * blockDim.x + threadIdx.x;
    if (t < n) out[t] = 0.0f;
}

__global__ void k_final_reduce(const float* __restrict__ H2,
                               const float* __restrict__ Wl,
                               const float* __restrict__ bl,
                               float* __restrict__ out) {
    __shared__ float partial[100];
    for (int j = threadIdx.x; j < 100; j += blockDim.x) partial[j] = 0.0f;
    __syncthreads();
    int n = blockIdx.x * blockDim.x + threadIdx.x;
    if (n < N_NODES) {
        float h[HEADS];
        #pragma unroll
        for (int i = 0; i < HEADS; ++i) h[i] = H2[(size_t)n * HEADS + i];
        for (int j = 0; j < 100; ++j) {
            float a = bl[j];
            #pragma unroll
            for (int i = 0; i < HEADS; ++i) a += h[i] * Wl[i * 100 + j];
            a = fmaxf(a, 0.0f);
            atomicAdd(&partial[j], a);            // LDS f32 atomic
        }
    }
    __syncthreads();
    for (int j = threadIdx.x; j < 100; j += blockDim.x)
        atomicAdd(out + j, partial[j] * (1.0f / (float)N_NODES));
}

// ---------------------------------------------------------------------------

extern "C" void kernel_launch(void* const* d_in, const int* in_sizes, int n_in,
                              void* d_out, int out_size, void* d_ws, size_t ws_size,
                              hipStream_t stream) {
    (void)in_sizes; (void)n_in; (void)out_size; (void)ws_size;

    const float* x  = (const float*)d_in[0];
    const int*   ei = (const int*)  d_in[1];
    const float* Wq1 = (const float*)d_in[2];  const float* bq1 = (const float*)d_in[3];
    const float* Wk1 = (const float*)d_in[4];  const float* bk1 = (const float*)d_in[5];
    const float* Wv1 = (const float*)d_in[6];  const float* bv1 = (const float*)d_in[7];
    const float* Ws1 = (const float*)d_in[8];  const float* bs1 = (const float*)d_in[9];
    const float* Wq2 = (const float*)d_in[10]; const float* bq2 = (const float*)d_in[11];
    const float* Wk2 = (const float*)d_in[12]; const float* bk2 = (const float*)d_in[13];
    const float* Wv2 = (const float*)d_in[14]; const float* bv2 = (const float*)d_in[15];
    const float* Ws2 = (const float*)d_in[16]; const float* bs2 = (const float*)d_in[17];
    const float* Wl  = (const float*)d_in[18]; const float* bl  = (const float*)d_in[19];
    float* out = (float*)d_out;

    // ---- workspace layout (256B aligned slots) ----
    size_t off = 0;
    auto alloc = [&](size_t bytes) {
        size_t o = off;
        off = (off + bytes + 255) & ~(size_t)255;
        return o;
    };
    char* ws = (char*)d_ws;
    __bf16* xbf   = (__bf16*)(ws + alloc((size_t)N_NODES * 128 * 2));
    __bf16* W1t   = (__bf16*)(ws + alloc((size_t)512 * 128 * 2));
    __bf16* W2t   = (__bf16*)(ws + alloc((size_t)32 * 128 * 2));
    float*  b1p   = (float*) (ws + alloc((size_t)512 * 4));
    float*  b2p   = (float*) (ws + alloc((size_t)32 * 4));
    float*  Q1    = (float*) (ws + alloc((size_t)N_NODES * 128 * 4));
    float*  K1    = (float*) (ws + alloc((size_t)N_NODES * 128 * 4));
    float*  V1    = (float*) (ws + alloc((size_t)N_NODES * 128 * 4));
    float*  H1    = (float*) (ws + alloc((size_t)N_NODES * 128 * 4)); // skip -> conv1 out
    __bf16* H1bf  = (__bf16*)(ws + alloc((size_t)N_NODES * 128 * 2));
    float*  Q2    = (float*) (ws + alloc((size_t)N_NODES * 8 * 4));
    float*  K2    = (float*) (ws + alloc((size_t)N_NODES * 8 * 4));
    float*  V2    = (float*) (ws + alloc((size_t)N_NODES * 8 * 4));
    float*  H2    = (float*) (ws + alloc((size_t)N_NODES * 8 * 4));   // skip -> conv2 out
    float*  scores= (float*) (ws + alloc((size_t)N_EDGES * HEADS * 4));
    unsigned* maxb= (unsigned*)(ws + alloc((size_t)N_NODES * HEADS * 4));
    float*  sumb  = (float*) (ws + alloc((size_t)N_NODES * HEADS * 4));

    const int EH       = N_EDGES * HEADS;                 // 6.4M
    const int gEH      = (EH + 255) / 256;
    const int gNH      = (N_NODES * HEADS + 255) / 256;
    const int gXcvt    = (N_NODES * 128 + 255) / 256;
    const int Mtiles   = N_NODES / 16;                    // 3125 exactly

    // ---- packing / conversion ----
    k_cvt_f32_to_bf16<<<gXcvt, 256, 0, stream>>>(x, xbf, N_NODES * 128);
    k_pack_weights<<<(512 * 128 + 255) / 256, 256, 0, stream>>>(
        Wq1, Wk1, Wv1, Ws1, bq1, bk1, bv1, bs1, W1t, b1p, 128, 128);
    k_pack_weights<<<(32 * 128 + 255) / 256, 256, 0, stream>>>(
        Wq2, Wk2, Wv2, Ws2, bq2, bk2, bv2, bs2, W2t, b2p, 128, 8);

    // ---- conv1: GEMMs (Q,K,V,skip) via WMMA ----
    k_wmma_gemm<128><<<dim3(Mtiles, 32), 32, 0, stream>>>(
        xbf, W1t, b1p, Q1, K1, V1, H1, 128);

    // ---- conv1: edge softmax + aggregation ----
    k_init_softmax<<<gNH, 256, 0, stream>>>(maxb, sumb, N_NODES * HEADS);
    k_score_max<16><<<gEH, 256, 0, stream>>>(ei, Q1, K1, scores, maxb, 0.25f);
    k_exp_sum<<<gEH, 256, 0, stream>>>(ei, maxb, scores, sumb);
    k_aggregate<16><<<gEH, 256, 0, stream>>>(ei, scores, sumb, V1, H1);

    // ---- conv2: GEMMs via WMMA ----
    k_cvt_f32_to_bf16<<<gXcvt, 256, 0, stream>>>(H1, H1bf, N_NODES * 128);
    k_wmma_gemm<8><<<dim3(Mtiles, 2), 32, 0, stream>>>(
        H1bf, W2t, b2p, Q2, K2, V2, H2, 128);

    // ---- conv2: edge softmax + aggregation (C = 1) ----
    k_init_softmax<<<gNH, 256, 0, stream>>>(maxb, sumb, N_NODES * HEADS);
    k_score_max<1><<<gEH, 256, 0, stream>>>(ei, Q2, K2, scores, maxb, 1.0f);
    k_exp_sum<<<gEH, 256, 0, stream>>>(ei, maxb, scores, sumb);
    k_aggregate<1><<<gEH, 256, 0, stream>>>(ei, scores, sumb, V2, H2);

    // ---- final head ----
    k_zero_out<<<1, 128, 0, stream>>>(out, 100);
    k_final_reduce<<<(N_NODES + 255) / 256, 256, 0, stream>>>(H2, Wl, bl, out);
}
